// ScaledDotProductAttention_60567628808550
// MI455X (gfx1250) — matrix operand order repaired and checked by reading, not compiled, not run on hardware
//
#include <hip/hip_runtime.h>

typedef __attribute__((ext_vector_type(16))) _Float16 v16h;
typedef __attribute__((ext_vector_type(2)))  __fp16   v2hf;   // cvt_pkrtz result type
typedef __attribute__((ext_vector_type(8)))  float    v8f;

#define S_LEN  2048
#define DHEAD  64
#define QTILE  128   // query rows per block (8 waves x 16)
#define KTILE  128   // keys per main-loop iteration (8 groups of 16)
#define NWAVES 8

// Scale folded into Q: sqrt(d)=8 (reference quirk) times log2(e) for base-2 softmax.
#define QSCALE 11.5415603272f   // 8 * 1.4426950408889634

// XOR-pattern lane shuffle via ds_swizzle_b32 (group-of-32 mode, immediate offset):
// offset[4:0]=and_mask(0x1f), offset[9:5]=or_mask(0), offset[14:10]=xor_mask.
#define SWZ_XOR(x, m) \
    __uint_as_float(__builtin_amdgcn_ds_swizzle(__float_as_uint(x), (0x1f | ((m) << 10))))

__global__ __launch_bounds__(256)
void fattn_f16_wmma_kernel(const float* __restrict__ qp,
                           const float* __restrict__ kp,
                           const float* __restrict__ vp,
                           float* __restrict__ op)
{
    // LDS: K tile row-major [key][d], V tile transposed [d][key], per-wave P staging (64 keys).
    __shared__ _Float16 Kt[KTILE * DHEAD];            // 16 KB
    __shared__ _Float16 Vt[DHEAD * KTILE];            // 16 KB
    __shared__ _Float16 Pt[NWAVES * 16 * 64];         // 16 KB (per-wave 16x64 staging)

    const int tid  = threadIdx.x;
    const int lane = tid & 31;
    const int wave = tid >> 5;
    const int m    = lane & 15;   // column/lane-in-half index
    const int hi   = lane >> 4;   // which 16-lane half

    const int bh = blockIdx.y;                     // 0..63  (B*H)
    const int qt = blockIdx.x;                     // 0..15  (query tile)
    const size_t base = (size_t)bh * S_LEN * DHEAD;

    const float* qg = qp + base;
    const float* kg = kp + base;
    const float* vg = vp + base;
    float*       og = op + base;

    // ---------------- Q A-fragments (16x32 f16 layout), scale folded in -------------------
    // A operand as loaded here (contiguous k, same as B): vgpr v holds halves {2v,2v+1}; K base = 2v + 16*hi
    const int qrow = qt * QTILE + wave * 16 + m;   // lane m owns row m of the wave tile
    union Frag { v16h h; v2hf p[8]; unsigned u[8]; };
    Frag qa[2];
#pragma unroll
    for (int c = 0; c < 2; ++c) {
#pragma unroll
        for (int vv = 0; vv < 8; ++vv) {
            const int kb = 2 * vv + hi * 16 + c * 32;
            const float2 f2 = *(const float2*)(qg + (size_t)qrow * DHEAD + kb);
            qa[c].p[vv] = __builtin_amdgcn_cvt_pkrtz(f2.x * QSCALE, f2.y * QSCALE);
        }
    }

    // ---------------- accumulators / online-softmax state (base-2 domain) -----------------
    v8f o[4];
#pragma unroll
    for (int g = 0; g < 4; ++g) o[g] = (v8f){0.f,0.f,0.f,0.f,0.f,0.f,0.f,0.f};
    float mrun[8], lrun[8];
#pragma unroll
    for (int r = 0; r < 8; ++r) { mrun[r] = -__builtin_inff(); lrun[r] = 0.0f; }

    _Float16* pw = &Pt[wave * 16 * 64];

    for (int kt0 = 0; kt0 < S_LEN; kt0 += KTILE) {
        __syncthreads();   // protect LDS tile from previous iteration's readers

        // ---- cooperative K/V tile load (f32 -> f16 packed), V stored transposed ----
#pragma unroll
        for (int i = 0; i < 8; ++i) {
            const int idx = tid + i * 256;     // float4 index, 0..2047
            const int r   = idx >> 4;          // key row 0..127
            const int c4  = (idx & 15) << 2;   // d column 0..60
            const float4 k4 = *(const float4*)(kg + (size_t)(kt0 + r) * DHEAD + c4);
            union { v2hf h; unsigned u; } p01, p23;
            p01.h = __builtin_amdgcn_cvt_pkrtz(k4.x, k4.y);
            p23.h = __builtin_amdgcn_cvt_pkrtz(k4.z, k4.w);
            *(uint2*)&Kt[r * DHEAD + c4] = make_uint2(p01.u, p23.u);   // one b64 store

            const float4 v4 = *(const float4*)(vg + (size_t)(kt0 + r) * DHEAD + c4);
            Vt[(c4 + 0) * KTILE + r] = (_Float16)v4.x;
            Vt[(c4 + 1) * KTILE + r] = (_Float16)v4.y;
            Vt[(c4 + 2) * KTILE + r] = (_Float16)v4.z;
            Vt[(c4 + 3) * KTILE + r] = (_Float16)v4.w;
        }
        // prefetch next tile into caches (global_prefetch_b8)
        if (kt0 + KTILE < S_LEN) {
            __builtin_prefetch(kg + (size_t)(kt0 + KTILE) * DHEAD + tid * 32, 0, 3);
            __builtin_prefetch(vg + (size_t)(kt0 + KTILE) * DHEAD + tid * 32, 0, 3);
        }
        __syncthreads();

        // ---- S = Q * K^T : 8 key groups of 16, K-dim 64 = 2 WMMAs each (16 WMMAs) ----
        // B operand as loaded here (contiguous k, same as A): lane n holds column n; halves h -> K row h + 16*hi.
        // B = K^T  =>  lane n reads key (g*16+n)'s d-values contiguously from Kt.
        v8f sfr[8];
#pragma unroll
        for (int g = 0; g < 8; ++g) {
            v8f acc = (v8f){0.f,0.f,0.f,0.f,0.f,0.f,0.f,0.f};
#pragma unroll
            for (int c = 0; c < 2; ++c) {
                const v16h bf = *(const v16h*)&Kt[(g * 16 + m) * DHEAD + c * 32 + hi * 16];
                acc = __builtin_amdgcn_wmma_f32_16x16x32_f16(
                        false, qa[c].h, false, bf, (short)0, acc, false, false);
            }
            sfr[g] = acc;
        }

        // ---- online softmax, base-2 (C layout: vgpr r -> row r + 8*hi, lane m -> col) ----
#pragma unroll
        for (int r = 0; r < 8; ++r) {
            float rowmax = fmaxf(fmaxf(fmaxf(sfr[0][r], sfr[1][r]), fmaxf(sfr[2][r], sfr[3][r])),
                                 fmaxf(fmaxf(sfr[4][r], sfr[5][r]), fmaxf(sfr[6][r], sfr[7][r])));
            rowmax = fmaxf(rowmax, SWZ_XOR(rowmax, 8));
            rowmax = fmaxf(rowmax, SWZ_XOR(rowmax, 4));
            rowmax = fmaxf(rowmax, SWZ_XOR(rowmax, 2));
            rowmax = fmaxf(rowmax, SWZ_XOR(rowmax, 1));
            const float mnew  = fmaxf(mrun[r], rowmax);
            const float alpha = __builtin_amdgcn_exp2f(mrun[r] - mnew);   // 2^(m_old-m_new)
            mrun[r] = mnew;
            float rsum = 0.0f;
#pragma unroll
            for (int g = 0; g < 8; ++g) {
                const float p = __builtin_amdgcn_exp2f(sfr[g][r] - mnew); // 2^(s-m)
                sfr[g][r] = p;
                rsum += p;
            }
            rsum += SWZ_XOR(rsum, 8);
            rsum += SWZ_XOR(rsum, 4);
            rsum += SWZ_XOR(rsum, 2);
            rsum += SWZ_XOR(rsum, 1);
            lrun[r] = lrun[r] * alpha + rsum;
#pragma unroll
            for (int g = 0; g < 4; ++g) o[g][r] *= alpha;
        }

        // ---- P*V in two 64-key phases through the per-wave staging buffer ----
        // (same-wave LDS ops are in-order: store->load->overwrite is safe without barriers)
#pragma unroll
        for (int half = 0; half < 2; ++half) {
            // stage P (C layout) -> LDS [m][key0..63]
#pragma unroll
            for (int g4 = 0; g4 < 4; ++g4)
#pragma unroll
                for (int r = 0; r < 8; ++r)
                    pw[(r + 8 * hi) * 64 + g4 * 16 + m] = (_Float16)sfr[half * 4 + g4][r];

            // reload in A layout
            Frag pa[2];
#pragma unroll
            for (int c = 0; c < 2; ++c)
#pragma unroll
                for (int vv = 0; vv < 8; ++vv) {
                    const int kb = 2 * vv + hi * 16 + c * 32;
                    pa[c].u[vv] = *(const unsigned*)&pw[m * 64 + kb];
                }

            // O += P * V : 4 d-column groups, 2 key chunks of 32 (8 WMMAs per phase)
            // B = V; Vt is [d][key] so lane n (col d = g*16+n) reads keys contiguously.
#pragma unroll
            for (int g = 0; g < 4; ++g)
#pragma unroll
                for (int c = 0; c < 2; ++c) {
                    const v16h vb = *(const v16h*)
                        &Vt[(g * 16 + m) * KTILE + half * 64 + c * 32 + hi * 16];
                    o[g] = __builtin_amdgcn_wmma_f32_16x16x32_f16(
                             false, pa[c].h, false, vb, (short)0, o[g], false, false);
                }
        }
    }

    // ---------------- epilogue: divide by l, coalesced f32 stores --------------------------
#pragma unroll
    for (int r = 0; r < 8; ++r) {
        const size_t orow = (size_t)qt * QTILE + wave * 16 + r + 8 * hi;
        const float inv = 1.0f / lrun[r];
#pragma unroll
        for (int g = 0; g < 4; ++g)
            og[orow * DHEAD + g * 16 + m] = o[g][r] * inv;
    }
}

extern "C" void kernel_launch(void* const* d_in, const int* in_sizes, int n_in,
                              void* d_out, int out_size, void* d_ws, size_t ws_size,
                              hipStream_t stream) {
    (void)n_in; (void)out_size; (void)d_ws; (void)ws_size;
    const float* q = (const float*)d_in[0];
    const float* k = (const float*)d_in[1];
    const float* v = (const float*)d_in[2];
    float* out = (float*)d_out;

    const int bh = in_sizes[0] / (S_LEN * DHEAD);   // B*H = 64
    dim3 grid(S_LEN / QTILE, bh);                   // 16 x 64 blocks
    fattn_f16_wmma_kernel<<<grid, 256, 0, stream>>>(q, k, v, out);
}
